// Informer_37701222924444
// MI455X (gfx1250) — compile-verified, matrix operand order checked by reading, not run on hardware
//
#include <hip/hip_runtime.h>
#include <hip/hip_bf16.h>
#include <math.h>

typedef __attribute__((ext_vector_type(16))) _Float16 v16h;
typedef __attribute__((ext_vector_type(8)))  _Float16 v8h;
typedef __attribute__((ext_vector_type(8)))  float    v8f;

#define TILE_M 128
#define TILE_N 128
#define TILE_K 64
#define LDSS   72   // LDS row stride in halves (144B, 16B aligned, pads banks)

// ---------------------------------------------------------------------------
// Tiled f16 WMMA GEMM:  C[M,N] = A[M,K] @ B[K,N] (+bias[n]) (+gelu)
// A,B row-major f16. Epilogue statically specialized via template params.
// M,N multiples of 128; K multiple of 64.
// Block = 256 threads = 8 waves; wave grid 4(m) x 2(n); each wave 32x64.
// 16 WMMAs per K-stage (two 16x16x32 sub-steps) between barriers.
// ---------------------------------------------------------------------------
template<bool HAS_BIAS, bool GELU, bool O32, bool O16>
__global__ __launch_bounds__(256)
void gemm_wmma_f16(const _Float16* __restrict__ A, const _Float16* __restrict__ Bm,
                   const float* __restrict__ bias,
                   float* __restrict__ out32, _Float16* __restrict__ out16,
                   int M, int N, int K)
{
    __shared__ __align__(16) _Float16 As[TILE_M * LDSS];
    __shared__ __align__(16) _Float16 Bs[TILE_N * LDSS];   // stored transposed: [n][k]

    const int tid  = threadIdx.x;
    const int lane = tid & 31;
    const int wave = tid >> 5;
    const int wm   = wave & 3;   // 0..3 -> 32 rows each
    const int wn   = wave >> 2;  // 0..1 -> 64 cols each
    const int m0   = blockIdx.y * TILE_M;
    const int n0   = blockIdx.x * TILE_N;

    v8f acc[2][4] = {};

    // staging assignments (TILE_K = 64)
    const int a_row = tid >> 1;          // 0..127
    const int a_kc  = (tid & 1) * 32;    // 0 or 32
    const int b_k   = tid >> 2;          // 0..63
    const int b_nc  = (tid & 3) * 32;    // 0,32,64,96

    const int kb  = (lane < 16) ? 0 : 8; // K sub-offset per WMMA 16-bit A/B layout
    const int r16 = lane & 15;

    for (int k0 = 0; k0 < K; k0 += TILE_K) {
        // ---- stage A tile (128 x 64), row-major in LDS ----
        {
            const v8h* src = (const v8h*)(A + (size_t)(m0 + a_row) * K + k0 + a_kc);
            v8h x0 = src[0], x1 = src[1], x2 = src[2], x3 = src[3];
            _Float16* dst = As + a_row * LDSS + a_kc;
            *(v8h*)(dst)      = x0;
            *(v8h*)(dst + 8)  = x1;
            *(v8h*)(dst + 16) = x2;
            *(v8h*)(dst + 24) = x3;
        }
        // ---- stage B tile (64 x 128) transposed into LDS as [n][k] ----
        {
            const v8h* src = (const v8h*)(Bm + (size_t)(k0 + b_k) * N + n0 + b_nc);
            v8h x0 = src[0], x1 = src[1], x2 = src[2], x3 = src[3];
            #pragma unroll
            for (int j = 0; j < 8; ++j) Bs[(b_nc + j)      * LDSS + b_k] = x0[j];
            #pragma unroll
            for (int j = 0; j < 8; ++j) Bs[(b_nc + 8 + j)  * LDSS + b_k] = x1[j];
            #pragma unroll
            for (int j = 0; j < 8; ++j) Bs[(b_nc + 16 + j) * LDSS + b_k] = x2[j];
            #pragma unroll
            for (int j = 0; j < 8; ++j) Bs[(b_nc + 24 + j) * LDSS + b_k] = x3[j];
        }
        // ---- prefetch next K-stage into cache hierarchy (global_prefetch) ----
        if (k0 + TILE_K < K) {
            __builtin_prefetch(A + (size_t)(m0 + a_row) * K + (k0 + TILE_K) + a_kc, 0, 3);
            __builtin_prefetch(Bm + (size_t)(k0 + TILE_K + b_k) * N + n0 + b_nc, 0, 3);
        }
        __syncthreads();

        // ---- two 16x16x32 K-substeps: 16 WMMAs per stage ----
        #pragma unroll
        for (int kk = 0; kk < TILE_K; kk += 32) {
            v16h afrag[2], bfrag[4];
            #pragma unroll
            for (int i = 0; i < 2; ++i) {
                const _Float16* p = As + (wm * 32 + i * 16 + r16) * LDSS + kk;
                v8h lo = *(const v8h*)(p + kb);        // K = kk+kb .. +7
                v8h hi = *(const v8h*)(p + kb + 16);   // K = kk+kb+16 .. +23
                #pragma unroll
                for (int j = 0; j < 8; ++j) { afrag[i][j] = lo[j]; afrag[i][j + 8] = hi[j]; }
            }
            #pragma unroll
            for (int j4 = 0; j4 < 4; ++j4) {
                const _Float16* p = Bs + (wn * 64 + j4 * 16 + r16) * LDSS + kk;
                v8h lo = *(const v8h*)(p + kb);
                v8h hi = *(const v8h*)(p + kb + 16);
                #pragma unroll
                for (int j = 0; j < 8; ++j) { bfrag[j4][j] = lo[j]; bfrag[j4][j + 8] = hi[j]; }
            }
            #pragma unroll
            for (int i = 0; i < 2; ++i)
                #pragma unroll
                for (int j = 0; j < 4; ++j)
                    acc[i][j] = __builtin_amdgcn_wmma_f32_16x16x32_f16(
                        false, afrag[i], false, bfrag[j], (short)0, acc[i][j], false, false);
        }
        __syncthreads();
    }

    // ---- epilogue: bias + optional gelu, statically-selected stores ----
    const int hi16 = lane >> 4;
    #pragma unroll
    for (int i = 0; i < 2; ++i) {
        #pragma unroll
        for (int j = 0; j < 4; ++j) {
            const int col = n0 + wn * 64 + j * 16 + r16;
            float bval = 0.f;
            if (HAS_BIAS) bval = bias[col];
            #pragma unroll
            for (int r = 0; r < 8; ++r) {
                const int row = m0 + wm * 32 + i * 16 + r + 8 * hi16;
                float v = acc[i][j][r] + bval;
                if (GELU) v = 0.5f * v * (1.f + erff(v * 0.70710678118654752f));
                if (O32) out32[(size_t)row * N + col] = v;
                if (O16) out16[(size_t)row * N + col] = (_Float16)v;
            }
        }
    }
}

// ---------------------------------------------------------------------------
// LayerNorm over rows of width 512, fused residual add.
// One wave per row; 8 rows per block. Writes f32 and/or f16 (nullable).
// ---------------------------------------------------------------------------
__global__ __launch_bounds__(256)
void ln512_kernel(const float* __restrict__ x, const float* __restrict__ res,
                  const float* __restrict__ g, const float* __restrict__ b,
                  float* __restrict__ y32, _Float16* __restrict__ y16, int rows)
{
    const int lane = threadIdx.x & 31;
    const int wave = threadIdx.x >> 5;
    const int row  = blockIdx.x * 8 + wave;
    if (row >= rows) return;

    const float* xr = x + (size_t)row * 512;
    const float* rr = res ? res + (size_t)row * 512 : nullptr;

    float v[16];
    float s = 0.f;
    #pragma unroll
    for (int j = 0; j < 16; ++j) {
        const int c = lane + j * 32;
        float t = xr[c];
        if (rr) t += rr[c];
        v[j] = t; s += t;
    }
    #pragma unroll
    for (int o = 16; o > 0; o >>= 1) s += __shfl_xor(s, o, 32);
    const float mean = s * (1.f / 512.f);

    float vs = 0.f;
    #pragma unroll
    for (int j = 0; j < 16; ++j) { const float d = v[j] - mean; vs += d * d; }
    #pragma unroll
    for (int o = 16; o > 0; o >>= 1) vs += __shfl_xor(vs, o, 32);
    const float inv = rsqrtf(vs * (1.f / 512.f) + 1e-5f);

    #pragma unroll
    for (int j = 0; j < 16; ++j) {
        const int c = lane + j * 32;
        const float o_ = (v[j] - mean) * inv * g[c] + b[c];
        if (y32) y32[(size_t)row * 512 + c] = o_;
        if (y16) y16[(size_t)row * 512 + c] = (_Float16)o_;
    }
}

// f32 -> f16 conversion (4 elements/thread, vectorized)
__global__ void cvt_f16_kernel(const float* __restrict__ s, _Float16* __restrict__ d, int n)
{
    const int i = (blockIdx.x * 256 + threadIdx.x) * 4;
    if (i + 3 < n) {
        float4 f = *(const float4*)(s + i);
        d[i] = (_Float16)f.x; d[i + 1] = (_Float16)f.y;
        d[i + 2] = (_Float16)f.z; d[i + 3] = (_Float16)f.w;
    } else {
        for (int j = i; j < n; ++j) d[j] = (_Float16)s[j];
    }
}

// b_attn[n] = bo[n] + sum_k bv[k] * wo[k*512 + n]   (512 outputs)
__global__ void battn_kernel(const float* __restrict__ bv, const float* __restrict__ wo,
                             const float* __restrict__ bo, float* __restrict__ out)
{
    const int n = blockIdx.x * 128 + threadIdx.x;
    if (n >= 512) return;
    float s = bo[n];
    for (int k = 0; k < 512; ++k) s += bv[k] * wo[k * 512 + n];
    out[n] = s;
}

// ---------------------------------------------------------------------------
extern "C" void kernel_launch(void* const* d_in, const int* in_sizes, int n_in,
                              void* d_out, int out_size, void* d_ws, size_t ws_size,
                              hipStream_t stream)
{
    const int D  = 512, DF = 2048, OUT = 256, EL = 3;
    const int M  = in_sizes[0] / D;        // 16384 rows (multiple of 128)

    // ---- inputs (setup_inputs order) ----
    const float* x       = (const float*)d_in[0];
    const float* w_in    = (const float*)d_in[1];
    const float* b_in    = (const float*)d_in[2];
    const float* wv      = (const float*)d_in[7];
    const float* bv      = (const float*)d_in[8];
    const float* wo      = (const float*)d_in[9];
    const float* bo      = (const float*)d_in[10];
    const float* conv1_w = (const float*)d_in[11];
    const float* conv1_b = (const float*)d_in[12];
    const float* conv2_w = (const float*)d_in[13];
    const float* conv2_b = (const float*)d_in[14];
    const float* n1_g    = (const float*)d_in[15];
    const float* n1_b    = (const float*)d_in[16];
    const float* n2_g    = (const float*)d_in[17];
    const float* n2_b    = (const float*)d_in[18];
    const float* norm_g  = (const float*)d_in[19];
    const float* norm_b  = (const float*)d_in[20];
    const float* w1      = (const float*)d_in[21];
    const float* b1      = (const float*)d_in[22];
    const float* w2      = (const float*)d_in[23];
    const float* b2      = (const float*)d_in[24];

    // ---- workspace carve-up ----
    size_t off = 0;
    auto a16 = [&](size_t elems) { _Float16* p = (_Float16*)((char*)d_ws + off);
                                   off += ((elems * 2 + 255) / 256) * 256; return p; };
    auto a32 = [&](size_t elems) { float* p = (float*)((char*)d_ws + off);
                                   off += ((elems * 4 + 255) / 256) * 256; return p; };

    _Float16* xf16    = a16((size_t)M * D);
    _Float16* hf16    = a16((size_t)M * D);
    _Float16* yf16    = a16((size_t)M * DF);
    float*    h32     = a32((size_t)M * D);
    float*    t32     = a32((size_t)M * D);
    _Float16* w_in16  = a16((size_t)D * D);
    _Float16* wv16    = a16((size_t)D * D);
    _Float16* wo16    = a16((size_t)D * D);
    _Float16* wattn16 = a16((size_t)D * D);
    _Float16* c1_16   = a16((size_t)EL * D * DF);
    _Float16* c2_16   = a16((size_t)EL * DF * D);
    _Float16* w1_16   = a16((size_t)D * DF);
    _Float16* w2_16   = a16((size_t)DF * OUT);
    float*    battn   = a32(D);
    (void)ws_size;

    auto cvt = [&](const float* s, _Float16* d, size_t n) {
        cvt_f16_kernel<<<dim3((unsigned)((n / 4 + 255) / 256)), 256, 0, stream>>>(s, d, (int)n);
    };
    auto grid = [&](int gm, int gn) { return dim3((unsigned)(gn / 128), (unsigned)(gm / 128)); };

    // ---- weight / input conversion to f16 ----
    cvt(x,       xf16,   (size_t)M * D);
    cvt(w_in,    w_in16, (size_t)D * D);
    cvt(wv,      wv16,   (size_t)D * D);
    cvt(wo,      wo16,   (size_t)D * D);
    cvt(conv1_w, c1_16,  (size_t)EL * D * DF);
    cvt(conv2_w, c2_16,  (size_t)EL * DF * D);
    cvt(w1,      w1_16,  (size_t)D * DF);
    cvt(w2,      w2_16,  (size_t)DF * OUT);

    // ---- fold attention (L==1 => attention output == V path):
    //      W_attn = wv @ wo ;  b_attn = bv @ wo + bo ----
    battn_kernel<<<dim3(4), 128, 0, stream>>>(bv, wo, bo, battn);
    gemm_wmma_f16<false, false, false, true><<<grid(D, D), 256, 0, stream>>>(
        wv16, wo16, nullptr, nullptr, wattn16, D, D, D);

    // ---- h = x @ w_in + b_in  (f32 master + f16 copy) ----
    gemm_wmma_f16<true, false, true, true><<<grid(M, D), 256, 0, stream>>>(
        xf16, w_in16, b_in, h32, hf16, M, D, D);

    // ---- encoder layers ----
    for (int i = 0; i < EL; ++i) {
        // a = h @ W_attn + b_attn
        gemm_wmma_f16<true, false, true, false><<<grid(M, D), 256, 0, stream>>>(
            hf16, wattn16, battn, t32, nullptr, M, D, D);
        // h = LN(h + a)
        ln512_kernel<<<dim3((unsigned)(M / 8)), 256, 0, stream>>>(
            h32, t32, n1_g + i * D, n1_b + i * D, h32, hf16, M);
        // y = gelu(h @ conv1 + b1) -> f16
        gemm_wmma_f16<true, true, false, true><<<grid(M, DF), 256, 0, stream>>>(
            hf16, c1_16 + (size_t)i * D * DF, conv1_b + i * DF, nullptr, yf16, M, DF, D);
        // t = y @ conv2 + b2
        gemm_wmma_f16<true, false, true, false><<<grid(M, D), 256, 0, stream>>>(
            yf16, c2_16 + (size_t)i * DF * D, conv2_b + i * D, t32, nullptr, M, D, DF);
        // h = LN(h + t)
        ln512_kernel<<<dim3((unsigned)(M / 8)), 256, 0, stream>>>(
            h32, t32, n2_g + i * D, n2_b + i * D, h32, hf16, M);
    }

    // ---- final norm + head ----
    ln512_kernel<<<dim3((unsigned)(M / 8)), 256, 0, stream>>>(
        h32, nullptr, norm_g, norm_b, nullptr, hf16, M);
    gemm_wmma_f16<true, true, false, true><<<grid(M, DF), 256, 0, stream>>>(
        hf16, w1_16, b1, nullptr, yf16, M, DF, D);
    gemm_wmma_f16<true, false, true, false><<<grid(M, OUT), 256, 0, stream>>>(
        yf16, w2_16, b2, (float*)d_out, nullptr, M, OUT, DF);
    (void)out_size; (void)n_in;
}
